// GCMC_50663434223642
// MI455X (gfx1250) — matrix operand order repaired and checked by reading, not compiled, not run on hardware
//
#include <hip/hip_runtime.h>
#include <math.h>

typedef __attribute__((ext_vector_type(16))) _Float16 v16h;
typedef __attribute__((ext_vector_type(2)))  __fp16   v2hf;  // cvt_pkrtz result type
typedef __attribute__((ext_vector_type(8)))  float    v8f;

static constexpr int kN     = 6144;   // particles
static constexpr int TILE   = 16;     // rows (i) per block = one WMMA M-tile
static constexpr int WAVES  = 8;      // waves per block, each owns a j-partition
static constexpr int NTHR   = WAVES * 32;
static constexpr int JCHUNK = 32;     // K per WMMA (f16)
static constexpr int JSPAN  = kN / WAVES;     // 768 j's per wave
static constexpr int NCHNK  = JSPAN / JCHUNK; // 24 WMMAs per wave
static constexpr float WSCALE = 4096.0f;  // 2^12 weight scale (cancels in alpha)

// ---------------------------------------------------------------------------
// Prep: V column-major f16, col d at Vcm[d*kN + j].
//   cols 0..3 = g, cols 4..7 = g^2, col 8 = 1.0 (softmax denom), col 9 = 0.
// g = [(s0+s2)/2, (s1+s3)/2, (s4+s6)/2, (s5+s7)/2]
// ---------------------------------------------------------------------------
__global__ void gcmc_prep(const float* __restrict__ states,
                          _Float16* __restrict__ Vcm) {
  int j = blockIdx.x * blockDim.x + threadIdx.x;
  if (j >= kN) return;
  float4 s0 = *(const float4*)(states + j * 8);
  float4 s1 = *(const float4*)(states + j * 8 + 4);
  float g0 = (s0.x + s0.z) * 0.5f;
  float g1 = (s0.y + s0.w) * 0.5f;
  float g2 = (s1.x + s1.z) * 0.5f;
  float g3 = (s1.y + s1.w) * 0.5f;
  Vcm[0 * kN + j] = (_Float16)g0;
  Vcm[1 * kN + j] = (_Float16)g1;
  Vcm[2 * kN + j] = (_Float16)g2;
  Vcm[3 * kN + j] = (_Float16)g3;
  Vcm[4 * kN + j] = (_Float16)(g0 * g0);
  Vcm[5 * kN + j] = (_Float16)(g1 * g1);
  Vcm[6 * kN + j] = (_Float16)(g2 * g2);
  Vcm[7 * kN + j] = (_Float16)(g3 * g3);
  Vcm[8 * kN + j] = (_Float16)1.0f;   // denominator column
  Vcm[9 * kN + j] = (_Float16)0.0f;   // padding column (N=9..15)
}

// ---------------------------------------------------------------------------
// Main: one block per 16-row tile. w @ [g, g^2, 1] via v_wmma_f32_16x16x32_f16.
// ---------------------------------------------------------------------------
__global__ void __launch_bounds__(NTHR)
gcmc_main(const float* __restrict__ states,
          const float* __restrict__ log_tau,
          const float* __restrict__ W1, const float* __restrict__ b1,
          const float* __restrict__ Wc, const float* __restrict__ bc,
          const float* __restrict__ Wu, const float* __restrict__ bu,
          const _Float16* __restrict__ Vcm,
          float* __restrict__ out) {
  __shared__ float s_posx[kN];                // 24 KB
  __shared__ float s_posy[kN];                // 24 KB
  __shared__ float s_red[WAVES * 256];        // 8 KB

  const int tid  = threadIdx.x;
  const int wave = tid >> 5;
  const int lane = tid & 31;
  const int row_base = blockIdx.x * TILE;

  // Stage positions into LDS (f32 for distance accuracy).
  for (int r = tid; r < kN; r += NTHR) {
    float4 s0 = *(const float4*)(states + r * 8);
    s_posx[r] = (s0.x + s0.z) * 0.5f;
    s_posy[r] = (s0.y + s0.w) * 0.5f;
  }
  __syncthreads();

  // w = exp(-dist/tau) * 2^12  (scale cancels in alpha = w/S; keeps f16 alive)
  const float tau = __builtin_amdgcn_exp2f(log_tau[0] * 1.44269504f);
  const float csc = 1.0f / (tau * 0.69314718056f);  // dist -> log2 units

  // A-operand K order used here (contiguous, matches B): lane<16 -> M=lane, K in {0..15};
  // lane>=16 -> M=lane-16, K in {16..31}.
  const int M     = lane & 15;
  const float pix = s_posx[row_base + M];
  const float piy = s_posy[row_base + M];
  const int kb    = (lane < 16) ? 0 : 16;  // A-operand K base
  const int koff  = (lane < 16) ? 0 : 16;  // B-operand K base
  // B column: features 0..7; col 8 = ones (denominator); cols 9..15 -> zeros.
  const int colidx = (M < 8) ? M : ((M == 8) ? 8 : 9);
  const _Float16* colbase = Vcm + colidx * kN + koff;

  v8f acc = {0.f, 0.f, 0.f, 0.f, 0.f, 0.f, 0.f, 0.f};

  int j0 = wave * JSPAN;
#pragma unroll 2
  for (int ch = 0; ch < NCHNK; ++ch, j0 += JCHUNK) {
    // ---- B operand: 16 consecutive f16 of this lane's column (32B aligned) ---
    const _Float16* colp = colbase + j0;
    __builtin_prefetch((const void*)(colp + JCHUNK), 0, 1);  // global_prefetch_b8
    v16h b = *(const v16h*)colp;   // 2x global_load_b128 (L2-resident)

    // ---- A operand: 16 weights, two runs of 8 consecutive j ----
    // (diagonal NOT masked here; exact self term 4096*V[i] subtracted at end)
    float wv[16];
    const int jb = j0 + kb;
#pragma unroll
    for (int run = 0; run < 2; ++run) {
      const int jr = jb + run * 8;
      float4 px0 = *(const float4*)(s_posx + jr);
      float4 px1 = *(const float4*)(s_posx + jr + 4);
      float4 py0 = *(const float4*)(s_posy + jr);
      float4 py1 = *(const float4*)(s_posy + jr + 4);
      float pxs[8] = {px0.x, px0.y, px0.z, px0.w, px1.x, px1.y, px1.z, px1.w};
      float pys[8] = {py0.x, py0.y, py0.z, py0.w, py1.x, py1.y, py1.z, py1.w};
#pragma unroll
      for (int k = 0; k < 8; ++k) {
        float dx = pix - pxs[k];
        float dy = piy - pys[k];
        float d  = __builtin_amdgcn_sqrtf(fmaf(dx, dx, dy * dy));
        wv[run * 8 + k] = __builtin_amdgcn_exp2f(fmaf(d, -csc, 12.0f));
      }
    }
    union { v16h v; v2hf p[8]; } au;
#pragma unroll
    for (int e = 0; e < 8; ++e)   // v_cvt_pk_rtz_f16_f32: 2 floats/op
      au.p[e] = __builtin_amdgcn_cvt_pkrtz(wv[2 * e], wv[2 * e + 1]);

    // D = A(16x32 f16 weights) x B(32x16 f16 values) + C  [f32 accum]
    acc = __builtin_amdgcn_wmma_f32_16x16x32_f16(
        false, au.v, false, b, (short)0, acc, false, false);
  }

  // ---- deterministic cross-wave reduction of 16x16 f32 accumulators ----
  float* myred = s_red + wave * 256;
#pragma unroll
  for (int r = 0; r < 8; ++r) myred[r * 32 + lane] = acc[r];
  __syncthreads();

  float sum = 0.0f;
#pragma unroll
  for (int w = 0; w < WAVES; ++w) sum += s_red[w * 256 + tid];
  s_red[tid] = sum;  // combined tile lives in wave-0 slot
  __syncthreads();

  // ---- finalize: 16 threads, one row each ----
  if (tid < TILE) {
    const int Mr = tid;
    const int i  = row_base + Mr;
    // C layout (ISA 7.12.2): M<8 -> vgpr=M, lane=N ; M>=8 -> vgpr=M-8, lane=N+16
    const int rr = (Mr < 8) ? Mr : (Mr - 8);
    const int lo = (Mr < 8) ? 0 : 16;
    auto comb = [&](int n) -> float { return s_red[rr * 32 + lo + n]; };

    float4 s0 = *(const float4*)(states + i * 8);
    float4 s1 = *(const float4*)(states + i * 8 + 4);
    float g[4] = {(s0.x + s0.z) * 0.5f, (s0.y + s0.w) * 0.5f,
                  (s1.x + s1.z) * 0.5f, (s1.y + s1.w) * 0.5f};

    // Remove the self term (w_ii = exp2(12-0) = 4096 exactly; V[i] rounded
    // through f16 exactly as the prep kernel produced it).
    const float S    = comb(8) - WSCALE;     // ones column
    const float invS = 1.0f / S;

    float f[8];
#pragma unroll
    for (int d = 0; d < 4; ++d) {
      float hg  = (float)(_Float16)g[d];             // f16(g_d), as stored
      float hg2 = (float)(_Float16)(g[d] * g[d]);    // f16(g_d^2), as stored
      float A = (comb(d)     - WSCALE * hg ) * invS; // (alpha @ g)_d
      float Q = (comb(4 + d) - WSCALE * hg2) * invS; // (alpha @ g^2)_d
      f[d]     = g[d] - A;                           // mu
      float vr = Q - A * A;                          // sigma^2
      f[4 + d] = __builtin_amdgcn_sqrtf(fmaxf(vr, 0.0f) + 1e-6f);
    }
#pragma unroll
    for (int d = 0; d < 8; ++d) out[4 * kN + i * 8 + d] = f[d];

    // MLP: h = relu(W1 f + b1); dv = Wc h + bc; sigma2 = softplus(Wu h + bu)+.01
    float h[16];
#pragma unroll
    for (int k = 0; k < 16; ++k) {
      float a1 = b1[k];
#pragma unroll
      for (int d = 0; d < 8; ++d) a1 = fmaf(W1[k * 8 + d], f[d], a1);
      h[k] = fmaxf(a1, 0.0f);
    }
#pragma unroll
    for (int o = 0; o < 2; ++o) {
      float ac = bc[o], au = bu[o];
#pragma unroll
      for (int k = 0; k < 16; ++k) {
        ac = fmaf(Wc[o * 16 + k], h[k], ac);
        au = fmaf(Wu[o * 16 + k], h[k], au);
      }
      out[i * 2 + o] = ac;                                    // dv
      float sp = (au > 20.0f) ? au : log1pf(expf(au));        // softplus
      out[2 * kN + i * 2 + o] = sp + 0.01f;                   // sigma2
    }
  }
}

// ---------------------------------------------------------------------------
extern "C" void kernel_launch(void* const* d_in, const int* in_sizes, int n_in,
                              void* d_out, int out_size, void* d_ws, size_t ws_size,
                              hipStream_t stream) {
  const float* states  = (const float*)d_in[0];
  const float* log_tau = (const float*)d_in[1];
  const float* W1 = (const float*)d_in[2];
  const float* b1 = (const float*)d_in[3];
  const float* Wc = (const float*)d_in[4];
  const float* bc = (const float*)d_in[5];
  const float* Wu = (const float*)d_in[6];
  const float* bu = (const float*)d_in[7];
  float* out = (float*)d_out;
  _Float16* Vcm = (_Float16*)d_ws;   // 10 * 6144 f16 = 120 KB scratch

  gcmc_prep<<<kN / 256, 256, 0, stream>>>(states, Vcm);
  gcmc_main<<<kN / TILE, NTHR, 0, stream>>>(states, log_tau, W1, b1, Wc, bc,
                                            Wu, bu, Vcm, out);
}
